// Retention_60258391162952
// MI455X (gfx1250) — matrix operand order repaired and checked by reading, not compiled, not run on hardware
//
#include <hip/hip_runtime.h>
#include <hip/hip_bf16.h>
#include <stdint.h>

#define LOG2_GAMMA -0.0458037f   // log2(0.96875)
#define Hd 1024
#define Ld 2048
#define Bd 4
#define BLr (Bd * Ld)            // 8192 flattened rows

typedef __attribute__((ext_vector_type(16))) __bf16 v16bf;
typedef __attribute__((ext_vector_type(8)))  __bf16 v8bf;
typedef __attribute__((ext_vector_type(8)))  float  v8f;
typedef __attribute__((ext_vector_type(4)))  unsigned int v4u;
typedef __attribute__((ext_vector_type(8)))  int v8i;
typedef __attribute__((ext_vector_type(4)))  int v4i;

// ---- WMMA / fragment helpers -----------------------------------------------

__device__ __forceinline__ v8f wmma_bf16(v16bf a, v16bf b, v8f c) {
    return __builtin_amdgcn_wmma_f32_16x16x32_bf16(false, a, false, b,
                                                   (short)0, c, false, false);
}

// 16-bit A fragment (16x32): lane holds row (lane&15); K chunks
// [16*hi, +8) and [16*hi+8, +8) relative to `row` -> two aligned 16B loads.
__device__ __forceinline__ v16bf load_a_frag(const __bf16* __restrict__ row, int hi) {
    v8bf lo  = *(const v8bf*)(row + 16 * hi);
    v8bf hi8 = *(const v8bf*)(row + 16 * hi + 8);
    v16bf a;
#pragma unroll
    for (int e = 0; e < 8; ++e) { a[e] = lo[e]; a[e + 8] = hi8[e]; }
    return a;
}

// 16-bit B fragment (32x16): lane holds column (lane&15), K values
// [16*hi, +16) contiguous -> one aligned 32B load.
__device__ __forceinline__ v16bf load_b_frag(const __bf16* __restrict__ col, int hi) {
    return *(const v16bf*)(col + 16 * hi);
}

// ---- CDNA5 async copy (ASYNCcnt path) ---------------------------------------

__device__ __forceinline__ uint32_t lds_addr_of(const void* p) {
    // flat->LDS mapping uses addr[31:0] (ISA 10.2): low bits are the LDS byte addr
    return (uint32_t)(uintptr_t)p;
}

__device__ __forceinline__ void async_copy16(uint32_t lds_byte, const void* gbase,
                                             uint32_t gbyte_off) {
    asm volatile("global_load_async_to_lds_b128 %0, %1, %2"
                 :: "v"(lds_byte), "v"(gbyte_off), "s"(gbase)
                 : "memory");
}

__device__ __forceinline__ void wait_async0() {
    asm volatile("s_wait_asynccnt 0x0" ::: "memory");
}

// ---- CDNA5 Tensor Data Mover (TENSORcnt path) -------------------------------
// Build a 2D D# (data_size=2B, row-major) and issue TENSOR_LOAD_TO_LDS.
// Tile [tile_h x tile_w] elements, tensor row stride `row_stride` elements,
// lands contiguously at lds_byte as [tile_h][tile_w] bf16.
__device__ __forceinline__ void tdm_load_2d(uint32_t lds_byte, const void* gptr,
                                            uint32_t tile_w, uint32_t tile_h,
                                            uint32_t tensor_w, uint32_t tensor_h,
                                            uint32_t row_stride) {
    uint64_t ga = (uint64_t)(uintptr_t)gptr;
    v4u g0;
    g0[0] = 1u;                                   // count=1, user descriptor
    g0[1] = lds_byte;                             // lds_addr
    g0[2] = (uint32_t)ga;                         // global_addr[31:0]
    g0[3] = (uint32_t)(ga >> 32) | (2u << 30);    // global_addr[56:32] | type=2
    v8i g1;
    g1[0] = (int)(1u << 16);                      // wg_mask=0, data_size=1 (2B)
    g1[1] = (int)((tensor_w & 0xFFFFu) << 16);    // tensor_dim0[15:0]
    g1[2] = (int)((tensor_w >> 16) | ((tensor_h & 0xFFFFu) << 16));
    g1[3] = (int)((tensor_h >> 16) | (tile_w << 16));   // dim1 hi | tile_dim0
    g1[4] = (int)(tile_h & 0xFFFFu);              // tile_dim1; tile_dim2=0
    g1[5] = (int)row_stride;                      // tensor_dim0_stride[31:0]
    g1[6] = 0;                                    // stride0 hi | dim1_stride lo
    g1[7] = 0;                                    // dim1_stride hi (unused, 2D)
    v4i g2 = {0, 0, 0, 0};                        // dims 2/3 unused
    v4i g3 = {0, 0, 0, 0};                        // dim 4 unused
    v8i g4 = {0, 0, 0, 0, 0, 0, 0, 0};            // extra group (unused here)
    __builtin_amdgcn_tensor_load_to_lds(g0, g1, g2, g3, g4, 0);
}

// ---- Precision conversion / layout kernels ---------------------------------

__global__ void cvt_f32_bf16(const float* __restrict__ src,
                             __bf16* __restrict__ dst, int n) {
    int i = blockIdx.x * blockDim.x + threadIdx.x;
    int stride = gridDim.x * blockDim.x;
    for (; i < n; i += stride) dst[i] = (__bf16)src[i];
}

// Wt[n*H + k] = (bf16) W[k*H + n]
__global__ void transpose_w_bf16(const float* __restrict__ w,
                                 __bf16* __restrict__ wt) {
    int idx = blockIdx.x * blockDim.x + threadIdx.x;   // H*H threads
    int k = idx >> 10, n = idx & (Hd - 1);
    wt[(size_t)n * Hd + k] = (__bf16)w[idx];
}

// ---- Shared GEMM tiling constants ------------------------------------------
#define MBLK 128
#define NBLK 64
#define KSTEP 32

// ---- Projection GEMM, TDM-staged (Q and K projections) ----------------------
__global__ __launch_bounds__(256) void proj_gemm_tdm(
        const __bf16* __restrict__ X, const __bf16* __restrict__ Wt,
        __bf16* __restrict__ out) {
    __shared__ __bf16 sA[2][MBLK * KSTEP];   // 2 x 8 KB
    __shared__ __bf16 sB[2][NBLK * KSTEP];   // 2 x 4 KB

    const int tid  = threadIdx.x;
    const int lane = tid & 31, wave = tid >> 5;
    const int hi = lane >> 4, nl = lane & 15;
    const int m0 = (blockIdx.x >> 4) * MBLK;
    const int n0 = (blockIdx.x & 15) * NBLK;
    const int wm = (wave >> 1) * 32, wn = (wave & 1) * 32;

    auto tdm_prefetch = [&](int p, int k0) {
        if (wave == 0) {      // wave-uniform branch: only wave 0 issues DMA
            tdm_load_2d(lds_addr_of(&sA[p][0]), X + (size_t)m0 * Hd + k0,
                        KSTEP, MBLK, Hd, BLr, Hd);
            tdm_load_2d(lds_addr_of(&sB[p][0]), Wt + (size_t)n0 * Hd + k0,
                        KSTEP, NBLK, Hd, Hd, Hd);
        }
    };

    v8f c[2][2] = {};
    tdm_prefetch(0, 0);
    const int STEPS = Hd / KSTEP;   // 32
    for (int it = 0; it < STEPS; ++it) {
        int p = it & 1;
        __builtin_amdgcn_s_wait_tensorcnt(0);  // wave0: DMA landed; others: NOP
        __syncthreads();                       // release all waves; reads done
        if (it + 1 < STEPS) tdm_prefetch(p ^ 1, (it + 1) * KSTEP);

        const __bf16* As = sA[p];
        const __bf16* Bs = sB[p];
        v16bf a[2], b[2];
#pragma unroll
        for (int mi = 0; mi < 2; ++mi)
            a[mi] = load_a_frag(As + (wm + 16 * mi + nl) * KSTEP, hi);
#pragma unroll
        for (int nj = 0; nj < 2; ++nj)
            b[nj] = load_b_frag(Bs + (wn + 16 * nj + nl) * KSTEP, hi);
#pragma unroll
        for (int mi = 0; mi < 2; ++mi)
#pragma unroll
            for (int nj = 0; nj < 2; ++nj)
                c[mi][nj] = wmma_bf16(a[mi], b[nj], c[mi][nj]);
    }

#pragma unroll
    for (int mi = 0; mi < 2; ++mi)
#pragma unroll
        for (int nj = 0; nj < 2; ++nj)
#pragma unroll
            for (int r = 0; r < 8; ++r) {
                int m = m0 + wm + 16 * mi + r + 8 * hi;
                int n = n0 + wn + 16 * nj + nl;
                out[(size_t)m * Hd + n] = (__bf16)c[mi][nj][r];
            }
}

// ---- Projection GEMM, async-LDS staged (V projection, transposed output) ----
__global__ __launch_bounds__(256) void proj_gemm_async(
        const __bf16* __restrict__ X, const __bf16* __restrict__ Wt,
        __bf16* __restrict__ out) {
    __shared__ __bf16 sA[2][MBLK * KSTEP];
    __shared__ __bf16 sB[2][NBLK * KSTEP];

    const int tid  = threadIdx.x;
    const int lane = tid & 31, wave = tid >> 5;
    const int hi = lane >> 4, nl = lane & 15;
    const int m0 = (blockIdx.x >> 4) * MBLK;
    const int n0 = (blockIdx.x & 15) * NBLK;
    const int wm = (wave >> 1) * 32, wn = (wave & 1) * 32;

    auto prefetch = [&](int p, int k0) {
#pragma unroll
        for (int r = 0; r < 2; ++r) {          // A: 512 x 16B chunks
            int cc = tid + 256 * r;
            int row = cc >> 2, kc = cc & 3;
            uint32_t goff = (uint32_t)(((m0 + row) * Hd + k0 + kc * 8) * 2);
            async_copy16(lds_addr_of(&sA[p][row * KSTEP + kc * 8]), X, goff);
        }
        {                                      // B: 256 x 16B chunks
            int row = tid >> 2, kc = tid & 3;
            uint32_t goff = (uint32_t)(((n0 + row) * Hd + k0 + kc * 8) * 2);
            async_copy16(lds_addr_of(&sB[p][row * KSTEP + kc * 8]), Wt, goff);
        }
    };

    v8f c[2][2] = {};
    prefetch(0, 0);
    const int STEPS = Hd / KSTEP;
    for (int it = 0; it < STEPS; ++it) {
        int p = it & 1;
        wait_async0();
        __syncthreads();
        if (it + 1 < STEPS) prefetch(p ^ 1, (it + 1) * KSTEP);

        const __bf16* As = sA[p];
        const __bf16* Bs = sB[p];
        v16bf a[2], b[2];
#pragma unroll
        for (int mi = 0; mi < 2; ++mi)
            a[mi] = load_a_frag(As + (wm + 16 * mi + nl) * KSTEP, hi);
#pragma unroll
        for (int nj = 0; nj < 2; ++nj)
            b[nj] = load_b_frag(Bs + (wn + 16 * nj + nl) * KSTEP, hi);
#pragma unroll
        for (int mi = 0; mi < 2; ++mi)
#pragma unroll
            for (int nj = 0; nj < 2; ++nj)
                c[mi][nj] = wmma_bf16(a[mi], b[nj], c[mi][nj]);
    }

    // V written per-batch transposed [H, L] for contiguous B-frags in S@V
#pragma unroll
    for (int mi = 0; mi < 2; ++mi)
#pragma unroll
        for (int nj = 0; nj < 2; ++nj)
#pragma unroll
            for (int r = 0; r < 8; ++r) {
                int m = m0 + wm + 16 * mi + r + 8 * hi;
                int n = n0 + wn + 16 * nj + nl;
                int b_ = m >> 11;          // batch
                int i  = m & (Ld - 1);     // row within batch
                out[(size_t)b_ * Hd * Ld + (size_t)n * Ld + i] =
                    (__bf16)c[mi][nj][r];
            }
}

// ---- Scores: S[i,j] = (Q_i . K_j) * gamma^(i-j) (causal), 32x32/wave -------
__global__ __launch_bounds__(256) void scores_gemm(
        const __bf16* __restrict__ Q, const __bf16* __restrict__ K,
        __bf16* __restrict__ S) {
    const int lane = threadIdx.x & 31, wave = threadIdx.x >> 5;
    const int hi = lane >> 4, nl = lane & 15;
    int wtile = blockIdx.x * 8 + wave;     // 64 x 64 = 4096 wave tiles
    int ti = wtile >> 6, tj = wtile & 63;
    int i0 = ti * 32, j0 = tj * 32;

    if (tj > ti) {                          // strictly above diagonal: zeros
#pragma unroll
        for (int mi = 0; mi < 2; ++mi)
#pragma unroll
            for (int nj = 0; nj < 2; ++nj)
#pragma unroll
                for (int r = 0; r < 8; ++r)
                    S[(size_t)(i0 + 16 * mi + r + 8 * hi) * Ld +
                      j0 + 16 * nj + nl] = (__bf16)0.0f;
        return;
    }

    v8f c[2][2] = {};
    for (int k0 = 0; k0 < Hd; k0 += 32) {
        v16bf a[2], b[2];
#pragma unroll
        for (int mi = 0; mi < 2; ++mi)
            a[mi] = load_a_frag(Q + (size_t)(i0 + 16 * mi + nl) * Hd + k0, hi);
#pragma unroll
        for (int nj = 0; nj < 2; ++nj)     // B[k=d][n=j] = K[j][d] contiguous
            b[nj] = load_b_frag(K + (size_t)(j0 + 16 * nj + nl) * Hd + k0, hi);
#pragma unroll
        for (int mi = 0; mi < 2; ++mi)
#pragma unroll
            for (int nj = 0; nj < 2; ++nj)
                c[mi][nj] = wmma_bf16(a[mi], b[nj], c[mi][nj]);
    }

#pragma unroll
    for (int mi = 0; mi < 2; ++mi)
#pragma unroll
        for (int nj = 0; nj < 2; ++nj)
#pragma unroll
            for (int r = 0; r < 8; ++r) {
                int i = i0 + 16 * mi + r + 8 * hi;
                int j = j0 + 16 * nj + nl;
                int d = i - j;
                float m = (d >= 0) ? exp2f((float)d * LOG2_GAMMA) : 0.0f;
                S[(size_t)i * Ld + j] = (__bf16)(c[mi][nj][r] * m);
            }
}

// ---- Output: O = S @ V (Vt stored [H,L]); causal K-truncation, 32x32/wave --
__global__ __launch_bounds__(256) void out_gemm(
        const __bf16* __restrict__ S, const __bf16* __restrict__ Vt,
        float* __restrict__ O) {
    const int lane = threadIdx.x & 31, wave = threadIdx.x >> 5;
    const int hi = lane >> 4, nl = lane & 15;
    int wtile = blockIdx.x * 8 + wave;     // 64 x 32 = 2048 wave tiles
    int ti = wtile >> 5, td = wtile & 31;
    int i0 = ti * 32, d0 = td * 32;

    int kmax = i0 + 32;                    // only j <= i0+31 contributes
    if (kmax > Ld) kmax = Ld;

    v8f c[2][2] = {};
    for (int k0 = 0; k0 < kmax; k0 += 32) {
        v16bf a[2], b[2];
#pragma unroll
        for (int mi = 0; mi < 2; ++mi)
            a[mi] = load_a_frag(S + (size_t)(i0 + 16 * mi + nl) * Ld + k0, hi);
#pragma unroll
        for (int nj = 0; nj < 2; ++nj)     // B[k=j][n=d] = V[j][d] = Vt[d][j]
            b[nj] = load_b_frag(Vt + (size_t)(d0 + 16 * nj + nl) * Ld + k0, hi);
#pragma unroll
        for (int mi = 0; mi < 2; ++mi)
#pragma unroll
            for (int nj = 0; nj < 2; ++nj)
                c[mi][nj] = wmma_bf16(a[mi], b[nj], c[mi][nj]);
    }

#pragma unroll
    for (int mi = 0; mi < 2; ++mi)
#pragma unroll
        for (int nj = 0; nj < 2; ++nj)
#pragma unroll
            for (int r = 0; r < 8; ++r)
                O[(size_t)(i0 + 16 * mi + r + 8 * hi) * Hd +
                  d0 + 16 * nj + nl] = c[mi][nj][r];
}

// ---- Host launcher ----------------------------------------------------------

extern "C" void kernel_launch(void* const* d_in, const int* in_sizes, int n_in,
                              void* d_out, int out_size, void* d_ws, size_t ws_size,
                              hipStream_t stream) {
    const float* x  = (const float*)d_in[0];
    const float* wq = (const float*)d_in[1];
    const float* wk = (const float*)d_in[2];
    const float* wv = (const float*)d_in[3];
    // d_in[4] (theta) is mathematically unused: thetas == ones.
    float* out = (float*)d_out;

    char* ws = (char*)d_ws;
    size_t off = 0;
    __bf16* xbf = (__bf16*)(ws + off); off += (size_t)BLr * Hd * 2;   // 16 MB
    __bf16* wqT = (__bf16*)(ws + off); off += (size_t)Hd * Hd * 2;    //  2 MB
    __bf16* wkT = (__bf16*)(ws + off); off += (size_t)Hd * Hd * 2;    //  2 MB
    __bf16* wvT = (__bf16*)(ws + off); off += (size_t)Hd * Hd * 2;    //  2 MB
    __bf16* qbf = (__bf16*)(ws + off); off += (size_t)BLr * Hd * 2;   // 16 MB
    __bf16* kbf = (__bf16*)(ws + off); off += (size_t)BLr * Hd * 2;   // 16 MB
    __bf16* vtb = (__bf16*)(ws + off); off += (size_t)BLr * Hd * 2;   // 16 MB
    __bf16* sbf = (__bf16*)(ws + off); off += (size_t)Ld * Ld * 2;    //  8 MB (reused)

    cvt_f32_bf16<<<4096, 256, 0, stream>>>(x, xbf, BLr * Hd);
    transpose_w_bf16<<<(Hd * Hd) / 256, 256, 0, stream>>>(wq, wqT);
    transpose_w_bf16<<<(Hd * Hd) / 256, 256, 0, stream>>>(wk, wkT);
    transpose_w_bf16<<<(Hd * Hd) / 256, 256, 0, stream>>>(wv, wvT);

    // (8192/128) x (1024/64) = 64 x 16 = 1024 workgroups
    proj_gemm_tdm<<<1024, 256, 0, stream>>>(xbf, wqT, qbf);        // TDM path
    proj_gemm_tdm<<<1024, 256, 0, stream>>>(xbf, wkT, kbf);        // TDM path
    proj_gemm_async<<<1024, 256, 0, stream>>>(xbf, wvT, vtb);      // async path

    for (int b = 0; b < Bd; ++b) {
        // stream order serializes reuse of sbf across batches
        scores_gemm<<<512, 256, 0, stream>>>(qbf + (size_t)b * Ld * Hd,
                                             kbf + (size_t)b * Ld * Hd, sbf);
        out_gemm<<<256, 256, 0, stream>>>(sbf, vtb + (size_t)b * Hd * 
                                          Ld, out + (size_t)b * Ld * Hd);
    }
}
